// LowRankCrossAttention_8581344657576
// MI455X (gfx1250) — compile-verified, matrix operand order checked by reading, not run on hardware
//
#include <hip/hip_runtime.h>

// ---------------------------------------------------------------------------
// LowRankCrossAttention on MI455X (gfx1250): all GEMMs via v_wmma_f32_16x16x32_bf16
// Pipeline (all in d_ws, bf16 storage / f32 accumulate):
//   cvt f32->bf16:  tgt, memory, Wq, Wk, Wv, Wo
//   q  = tgt  . Wq^T * SCALE        [B*T, 128]   (bf16)
//   k  = mem  . Wk^T                [B*S, 128]   (bf16)
//   vT = (mem . Wv^T)^T             [D, B*S]     (bf16, transposed store)
//   sc = q . k^T  (batched)         [B, T, S]    (f32)
//   at = softmax_rows(sc)           [B, T, S]    (bf16)
//   ao = at . vT^T (batched)        [B, T, D]    (bf16)
//   out= ao . Wo^T                  [B*T, D]     (f32 -> d_out)
// GEMM: 128-thread blocks = 4 waves, each wave owns a 64x64 tile -> 16 WMMAs
// per K-step against 16 b128 fragment loads (1:1 math:load instruction ratio).
// ---------------------------------------------------------------------------

typedef __bf16 bf16;
typedef bf16  v16bf __attribute__((ext_vector_type(16)));
typedef bf16  v8bf  __attribute__((ext_vector_type(8)));
typedef bf16  v4bf  __attribute__((ext_vector_type(4)));
typedef float v8f   __attribute__((ext_vector_type(8)));

// Load one bf16 WMMA operand fragment (16x32 A or 32x16 B tile), row-major
// source with contraction contiguous. Per ISA 7.12.2 (16-bit, K=32):
//   lanes 0-15  hold K in [0,8) and [16,24)
//   lanes 16-31 hold K in [8,16) and [24,32)
// => two 16-byte loads per lane.
__device__ __forceinline__ v16bf load_frag16(const bf16* __restrict__ p, int hi) {
  union { v16bf f; v8bf h[2]; } u;
  u.h[0] = *(const v8bf*)(p + hi * 8);
  u.h[1] = *(const v8bf*)(p + 16 + hi * 8);
  return u.f;
}

// Generic bf16 GEMM: C[m,n] = scale * sum_k A[m,k] * Bm[n,k]
// A: [*, lda] bf16 (contraction contiguous), Bm: [*, ldb] bf16 (contraction contiguous)
// OUTM: 0 = f32 C[m*ldc+n], 1 = bf16 C[m*ldc+n], 2 = bf16 transposed C[n*ldc+m]
// grid = (M/128, N/128, batch), block = 128 (4 waves, 2x2 wave grid, 64x64/wave)
template <int OUTM>
__global__ __launch_bounds__(128) void wmma_gemm(
    const bf16* __restrict__ A, const bf16* __restrict__ Bm, void* __restrict__ Cv,
    int K, int lda, int ldb, int ldc,
    long long sA, long long sB, long long sC, float scale) {
  const int tid   = threadIdx.x;
  const int lane  = tid & 31;
  const int wave  = tid >> 5;
  const int laneL = lane & 15;
  const int hi    = lane >> 4;
  const int wm    = wave & 1;       // 2 wave-rows of 64
  const int wn    = wave >> 1;      // 2 wave-cols of 64
  const int rowBase = blockIdx.x * 128 + wm * 64;
  const int colBase = blockIdx.y * 128 + wn * 64;
  const int bz = blockIdx.z;

  const bf16* __restrict__ Ab = A  + (long long)bz * sA;
  const bf16* __restrict__ Bb = Bm + (long long)bz * sB;

  v8f acc[4][4];
#pragma unroll
  for (int mi = 0; mi < 4; ++mi)
#pragma unroll
    for (int ni = 0; ni < 4; ++ni)
#pragma unroll
      for (int r = 0; r < 8; ++r) acc[mi][ni][r] = 0.0f;

  // Per-lane row pointers (include the lane's hi-half K offset handled inside
  // load_frag16). All row indices are in-bounds: every dim is a multiple of 128.
  const bf16* aPtr[4];
#pragma unroll
  for (int mi = 0; mi < 4; ++mi)
    aPtr[mi] = Ab + (long long)(rowBase + mi * 16 + laneL) * lda;
  const bf16* bPtr[4];
#pragma unroll
  for (int ni = 0; ni < 4; ++ni)
    bPtr[ni] = Bb + (long long)(colBase + ni * 16 + laneL) * ldb;

  for (int k0 = 0; k0 < K; k0 += 32) {
    v16bf aF[4];
#pragma unroll
    for (int mi = 0; mi < 4; ++mi) aF[mi] = load_frag16(aPtr[mi] + k0, hi);
    v16bf bF[4];
#pragma unroll
    for (int ni = 0; ni < 4; ++ni) bF[ni] = load_frag16(bPtr[ni] + k0, hi);
#pragma unroll
    for (int mi = 0; mi < 4; ++mi)
#pragma unroll
      for (int ni = 0; ni < 4; ++ni)
        acc[mi][ni] = __builtin_amdgcn_wmma_f32_16x16x32_bf16(
            false, aF[mi], false, bF[ni], (short)0, acc[mi][ni], false, false);
  }

  // Epilogue. C/D layout: element (m = r + hi*8, n = laneL) in f32 VGPR r.
#pragma unroll
  for (int mi = 0; mi < 4; ++mi) {
#pragma unroll
    for (int ni = 0; ni < 4; ++ni) {
      const int mB = rowBase + mi * 16 + hi * 8;
      const int n  = colBase + ni * 16 + laneL;
      if (OUTM == 0) {
        float* C = (float*)Cv + (long long)bz * sC;
#pragma unroll
        for (int r = 0; r < 8; ++r)
          C[(long long)(mB + r) * ldc + n] = acc[mi][ni][r] * scale;
      } else if (OUTM == 1) {
        bf16* C = (bf16*)Cv + (long long)bz * sC;
#pragma unroll
        for (int r = 0; r < 8; ++r)
          C[(long long)(mB + r) * ldc + n] = (bf16)(acc[mi][ni][r] * scale);
      } else {
        bf16* C = (bf16*)Cv + (long long)bz * sC;
        v8bf pk;
#pragma unroll
        for (int r = 0; r < 8; ++r) pk[r] = (bf16)(acc[mi][ni][r] * scale);
        *(v8bf*)(C + (long long)n * ldc + mB) = pk;   // contiguous 16B store
      }
    }
  }
}

// Vectorized f32 -> bf16 conversion (4 elems / thread)
__global__ __launch_bounds__(256) void cvt_f32_bf16(
    const float* __restrict__ in, bf16* __restrict__ out, long long n4) {
  long long i = (long long)blockIdx.x * 256 + threadIdx.x;
  if (i < n4) {
    float4 f = ((const float4*)in)[i];
    v4bf o;
    o[0] = (bf16)f.x; o[1] = (bf16)f.y; o[2] = (bf16)f.z; o[3] = (bf16)f.w;
    ((v4bf*)out)[i] = o;
  }
}

// Row softmax over S=2048, one 256-thread block per row, 8 elems cached in regs.
__global__ __launch_bounds__(256) void softmax_rows(
    const float* __restrict__ scores, bf16* __restrict__ attn) {
  const int Scols = 2048;
  const long long row = blockIdx.x;
  const float* __restrict__ src = scores + row * Scols;
  bf16* __restrict__ dst = attn + row * Scols;
  const int tid = threadIdx.x;
  const int lane = tid & 31, wave = tid >> 5;

  float v[8];
  float lmax = -3.402823466e38f;
#pragma unroll
  for (int i = 0; i < 8; ++i) {
    v[i] = src[tid + (i << 8)];
    lmax = fmaxf(lmax, v[i]);
  }
  __shared__ float red[8];
#pragma unroll
  for (int o = 16; o > 0; o >>= 1) lmax = fmaxf(lmax, __shfl_down(lmax, o, 32));
  if (lane == 0) red[wave] = lmax;
  __syncthreads();
  float m = red[0];
#pragma unroll
  for (int w = 1; w < 8; ++w) m = fmaxf(m, red[w]);
  __syncthreads();

  float lsum = 0.0f;
#pragma unroll
  for (int i = 0; i < 8; ++i) {
    v[i] = __expf(v[i] - m);
    lsum += v[i];
  }
#pragma unroll
  for (int o = 16; o > 0; o >>= 1) lsum += __shfl_down(lsum, o, 32);
  if (lane == 0) red[wave] = lsum;
  __syncthreads();
  float tot = red[0];
#pragma unroll
  for (int w = 1; w < 8; ++w) tot += red[w];
  const float inv = 1.0f / tot;
#pragma unroll
  for (int i = 0; i < 8; ++i) dst[tid + (i << 8)] = (bf16)(v[i] * inv);
}

extern "C" void kernel_launch(void* const* d_in, const int* in_sizes, int n_in,
                              void* d_out, int out_size, void* d_ws, size_t ws_size,
                              hipStream_t stream) {
  constexpr int B = 4, T = 2048, S = 2048, D = 1024, R = 128;
  constexpr float SCALE = 0.08838834764831845f;  // 1/sqrt(128)
  (void)in_sizes; (void)n_in; (void)out_size; (void)ws_size;

  const float* tgt = (const float*)d_in[0];
  const float* mem = (const float*)d_in[1];
  const float* Wq  = (const float*)d_in[2];
  const float* Wk  = (const float*)d_in[3];
  const float* Wv  = (const float*)d_in[4];
  const float* Wo  = (const float*)d_in[5];

  char* base = (char*)d_ws;
  size_t off = 0;
  auto take = [&](size_t bytes) -> char* {
    char* p = base + off;
    off = (off + bytes + 255) & ~(size_t)255;
    return p;
  };
  bf16*  tgt_b = (bf16*)take((size_t)B * T * D * 2);
  bf16*  mem_b = (bf16*)take((size_t)B * S * D * 2);
  bf16*  Wq_b  = (bf16*)take((size_t)R * D * 2);
  bf16*  Wk_b  = (bf16*)take((size_t)R * D * 2);
  bf16*  Wv_b  = (bf16*)take((size_t)D * D * 2);
  bf16*  Wo_b  = (bf16*)take((size_t)D * D * 2);
  bf16*  q_b   = (bf16*)take((size_t)B * T * R * 2);
  bf16*  k_b   = (bf16*)take((size_t)B * S * R * 2);
  bf16*  vT_b  = (bf16*)take((size_t)D * B * S * 2);   // [D, B*S]
  float* sc_f  = (float*)take((size_t)B * T * S * 4);
  bf16*  at_b  = (bf16*)take((size_t)B * T * S * 2);
  bf16*  ao_b  = (bf16*)take((size_t)B * T * D * 2);

  const dim3 blkG(128);   // GEMM: 4 waves
  const dim3 blkE(256);   // elementwise / softmax
  auto cvt = [&](const float* in, bf16* out, long long n) {
    long long n4 = n / 4;
    cvt_f32_bf16<<<dim3((unsigned)((n4 + 255) / 256)), blkE, 0, stream>>>(in, out, n4);
  };
  cvt(tgt, tgt_b, (long long)B * T * D);
  cvt(mem, mem_b, (long long)B * S * D);
  cvt(Wq,  Wq_b,  (long long)R * D);
  cvt(Wk,  Wk_b,  (long long)R * D);
  cvt(Wv,  Wv_b,  (long long)D * D);
  cvt(Wo,  Wo_b,  (long long)D * D);

  // q = tgt . Wq^T * SCALE   [8192 x 128], K=1024
  wmma_gemm<1><<<dim3(64, 1, 1), blkG, 0, stream>>>(
      tgt_b, Wq_b, q_b, D, D, D, R, 0, 0, 0, SCALE);
  // k = mem . Wk^T           [8192 x 128], K=1024
  wmma_gemm<1><<<dim3(64, 1, 1), blkG, 0, stream>>>(
      mem_b, Wk_b, k_b, D, D, D, R, 0, 0, 0, 1.0f);
  // vT = (mem . Wv^T)^T      stored [1024 x 8192], K=1024
  wmma_gemm<2><<<dim3(64, 8, 1), blkG, 0, stream>>>(
      mem_b, Wv_b, vT_b, D, D, D, B * S, 0, 0, 0, 1.0f);
  // scores = q . k^T (batched) [4][2048 x 2048], K=128 (SCALE already in q)
  wmma_gemm<0><<<dim3(16, 16, 4), blkG, 0, stream>>>(
      q_b, k_b, sc_f, R, R, R, S,
      (long long)T * R, (long long)S * R, (long long)T * S, 1.0f);
  // attn = softmax(scores)
  softmax_rows<<<dim3(B * T), blkE, 0, stream>>>(sc_f, at_b);
  // ao = attn . v (batched) [4][2048 x 1024], K=2048; B-operand = vT rows
  wmma_gemm<1><<<dim3(16, 8, 4), blkG, 0, stream>>>(
      at_b, vT_b, ao_b, S, S, B * S, D,
      (long long)T * S, (long long)S, (long long)T * D, 1.0f);
  // out = ao . Wo^T          [8192 x 1024] f32, K=1024
  wmma_gemm<0><<<dim3(64, 8, 1), blkG, 0, stream>>>(
      ao_b, Wo_b, d_out, D, D, D, D, 0, 0, 0, 1.0f);
}
